// LSTM_21285857919546
// MI455X (gfx1250) — matrix operand order repaired and checked by reading, not compiled, not run on hardware
//
#include <hip/hip_runtime.h>
#include <hip/hip_bf16.h>
#include <math.h>

typedef __bf16 bf16_t;
typedef __attribute__((ext_vector_type(16))) __bf16 v16bf;
typedef __attribute__((ext_vector_type(8)))  __bf16 v8bf;
typedef __attribute__((ext_vector_type(8)))  float  v8f;

#define H        1500
#define G4       6000          // 4*H gate rows
#define IN_SZ    20
#define T_STEPS  4096
#define KPAD     1536          // H padded to multiple of 32
#define KT       48            // KPAD / 32
#define MLP_H    1875
#define OUT_N    20

#define NBLK     125           // persistent blocks
#define WPB      3             // waves per block -> 125*3 = 375 tiles = 6000/16
#define BDIM     (WPB * 32)

// ---- workspace layout (bytes) ----
#define WS_CNT   0             // int   grid-barrier arrival counter
#define WS_GEN   4             // int   grid-barrier generation
#define WS_HBF   64            // bf16[1536]  hidden state (bf16, zero-padded tail)
#define WS_HF    3200          // float[1500] hidden state (fp32, for MLP)
#define WS_HID   9216          // float[1875] MLP hidden
#define WS_WPK   16768         // bf16[KT*G4*32] packed W_hh  (~18.4 MB)
#define WS_ZERO_BYTES 9216     // must zero cnt/gen/hbf/hf every launch

// Pack W_hh into WMMA-native bf16 tiles.
// Column order ncol = 4*j + g  (4 gates of hidden unit j adjacent).
// For each (kt, ncol): 32 consecutive bf16 = K values kt*32 .. kt*32+31.
__global__ void pack_whh(const float* __restrict__ W_hh, bf16_t* __restrict__ wp) {
    int idx = blockIdx.x * blockDim.x + threadIdx.x;
    const int total = KT * G4 * 32;
    if (idx >= total) return;
    int kk   = idx & 31;
    int ncol = (idx >> 5) % G4;
    int kt   = idx / (32 * G4);
    int k    = kt * 32 + kk;
    int j    = ncol >> 2;
    int g    = ncol & 3;
    float v  = 0.0f;
    if (k < H) v = W_hh[(size_t)(g * H + j) * H + k];
    wp[idx] = (bf16_t)v;
}

// Persistent LSTM scan: one wave owns 4 hidden units (16 gate columns).
// gates tile via v_wmma_f32_16x16x32_bf16; grid barrier each timestep.
__global__ void __launch_bounds__(BDIM) lstm_scan(
    const float*  __restrict__ in,
    const float*  __restrict__ W_ih,
    const float*  __restrict__ b_ih,
    const float*  __restrict__ b_hh,
    const bf16_t* __restrict__ wpack,
    bf16_t* hbf, float* hf, int* cnt, int* gen)
{
    __shared__ alignas(64) bf16_t hsh[KPAD];

    const int lane = threadIdx.x & 31;
    const int wid  = threadIdx.x >> 5;
    const int tile = blockIdx.x * WPB + wid;   // 0..374
    const int lh   = lane >> 4;                // lane half
    const int n    = lane & 15;                // column within tile
    const int ncol = tile * 16 + n;            // 0..5999
    const int g    = ncol & 3;                 // gate: 0=i 1=f 2=g 3=o
    const int j    = ncol >> 2;                // hidden unit 0..1499
    const int row  = g * H + j;                // row of W_ih / biases

    const float bias = b_ih[row] + b_hh[row];
    float wih[IN_SZ];
#pragma unroll
    for (int k = 0; k < IN_SZ; ++k) wih[k] = W_ih[row * IN_SZ + k];

    float cst = 0.0f;                          // c state, register-resident

    for (int t = 0; t < T_STEPS; ++t) {
        // stage h (bf16, padded) into LDS
        {
            const unsigned* src = (const unsigned*)hbf;
            unsigned*       dst = (unsigned*)hsh;
            for (int k = threadIdx.x; k < KPAD / 2; k += BDIM) dst[k] = src[k];
        }
        __syncthreads();

        // gates[ncol] partial = sum_k W_hh[row][k] * h[k]  via WMMA
        v8f acc = {0.f, 0.f, 0.f, 0.f, 0.f, 0.f, 0.f, 0.f};
#pragma unroll 4
        for (int kt = 0; kt < KT; ++kt) {
            // A: h broadcast across all 16 rows. Contiguous K order to match B:
            // lh=0 holds K {0..15}, lh=1 holds K {16..31}.
            v8bf alo = *(const v8bf*)&hsh[kt * 32 + lh * 16];
            v8bf ahi = *(const v8bf*)&hsh[kt * 32 + lh * 16 + 8];
            v16bf a = __builtin_shufflevector(alo, ahi,
                        0,1,2,3,4,5,6,7,8,9,10,11,12,13,14,15);
            // B: per-lane contiguous 16 K-values (lanes 0-15: K 0..15, 16-31: K 16..31)
            const v16bf* bp = (const v16bf*)
                &wpack[((size_t)kt * G4 + (size_t)tile * 16 + n) * 32 + lh * 16];
            acc = __builtin_amdgcn_wmma_f32_16x16x32_bf16(
                      false, a, false, *bp, (short)0, acc, false, false);
        }
        float y = acc[0];   // all D rows identical (A rows identical)

        // fold in input projection for this timestep (W_ih row in registers)
        const float* xr = in + (size_t)t * IN_SZ;
        float xd = 0.0f;
#pragma unroll
        for (int k = 0; k < IN_SZ; ++k) xd += xr[k] * wih[k];
        float gate = y + xd + bias;

        // gather the 4 gates of this hidden unit from adjacent lanes
        int base = lane & ~3;
        float gi = __shfl(gate, base + 0, 32);
        float gf = __shfl(gate, base + 1, 32);
        float gg = __shfl(gate, base + 2, 32);
        float go = __shfl(gate, base + 3, 32);
        gi = 1.0f / (1.0f + __expf(-gi));
        gf = 1.0f / (1.0f + __expf(-gf));
        gg = tanhf(gg);
        go = 1.0f / (1.0f + __expf(-go));
        cst = gf * cst + gi * gg;
        float hnew = go * tanhf(cst);

        if (lane < 16 && (lane & 3) == 0) {
            hbf[j] = (bf16_t)hnew;
            hf[j]  = hnew;
        }

        // ---- grid-wide sense/generation barrier ----
        __syncthreads();
        if (threadIdx.x == 0) {
            __threadfence();
            int prev = __hip_atomic_fetch_add(cnt, 1, __ATOMIC_ACQ_REL,
                                              __HIP_MEMORY_SCOPE_AGENT);
            if (prev == NBLK - 1) {
                __hip_atomic_store(cnt, 0, __ATOMIC_RELAXED,
                                   __HIP_MEMORY_SCOPE_AGENT);
                __hip_atomic_store(gen, t + 1, __ATOMIC_RELEASE,
                                   __HIP_MEMORY_SCOPE_AGENT);
            } else {
                while (__hip_atomic_load(gen, __ATOMIC_ACQUIRE,
                                         __HIP_MEMORY_SCOPE_AGENT) <= t) {
                    __builtin_amdgcn_s_sleep(1);
                }
            }
        }
        __syncthreads();
    }
}

// hid = W1 @ h_last + b1   (wave per row, butterfly reduce)
__global__ void mlp1(const float* __restrict__ W1, const float* __restrict__ b1,
                     const float* __restrict__ hf, float* __restrict__ hid) {
    int r    = blockIdx.x * 8 + (threadIdx.x >> 5);
    int lane = threadIdx.x & 31;
    if (r >= MLP_H) return;
    float acc = 0.0f;
    for (int k = lane; k < H; k += 32) acc += W1[(size_t)r * H + k] * hf[k];
#pragma unroll
    for (int o = 16; o; o >>= 1) acc += __shfl_xor(acc, o, 32);
    if (lane == 0) hid[r] = acc + b1[r];
}

// out = W2 @ hid + b2   (20 waves in one block)
__global__ void mlp2(const float* __restrict__ W2, const float* __restrict__ b2,
                     const float* __restrict__ hid, float* __restrict__ out) {
    int r    = threadIdx.x >> 5;
    int lane = threadIdx.x & 31;
    float acc = 0.0f;
    for (int k = lane; k < MLP_H; k += 32) acc += W2[(size_t)r * MLP_H + k] * hid[k];
#pragma unroll
    for (int o = 16; o; o >>= 1) acc += __shfl_xor(acc, o, 32);
    if (lane == 0) out[r] = acc + b2[r];
}

extern "C" void kernel_launch(void* const* d_in, const int* in_sizes, int n_in,
                              void* d_out, int out_size, void* d_ws, size_t ws_size,
                              hipStream_t stream) {
    (void)in_sizes; (void)n_in; (void)out_size; (void)ws_size;
    const float* in   = (const float*)d_in[0];
    const float* W_ih = (const float*)d_in[1];
    const float* W_hh = (const float*)d_in[2];
    const float* b_ih = (const float*)d_in[3];
    const float* b_hh = (const float*)d_in[4];
    const float* W1   = (const float*)d_in[5];
    const float* b1   = (const float*)d_in[6];
    const float* W2   = (const float*)d_in[7];
    const float* b2   = (const float*)d_in[8];

    char*   ws  = (char*)d_ws;
    int*    cnt = (int*)(ws + WS_CNT);
    int*    gen = (int*)(ws + WS_GEN);
    bf16_t* hbf = (bf16_t*)(ws + WS_HBF);
    float*  hf  = (float*)(ws + WS_HF);
    float*  hid = (float*)(ws + WS_HID);
    bf16_t* wpk = (bf16_t*)(ws + WS_WPK);

    // deterministic reset of barrier state + h0 = c0 = 0 (capture-safe)
    hipMemsetAsync(d_ws, 0, WS_ZERO_BYTES, stream);

    const int total = KT * G4 * 32;
    pack_whh<<<(total + 255) / 256, 256, 0, stream>>>(W_hh, wpk);
    lstm_scan<<<NBLK, BDIM, 0, stream>>>(in, W_ih, b_ih, b_hh, wpk,
                                         hbf, hf, cnt, gen);
    mlp1<<<(MLP_H + 7) / 8, 256, 0, stream>>>(W1, b1, hf, hid);
    mlp2<<<1, 640, 0, stream>>>(W2, b2, hid, (float*)d_out);
}
